// SAGE3BN3MLP2_noGlobalMeanPool_61426622267897
// MI455X (gfx1250) — matrix operand order repaired and checked by reading, not compiled, not run on hardware
//
#include <hip/hip_runtime.h>
#include <math.h>

#define HD   128          // hidden dim
#define CD   40           // classes
#define CDP  48           // classes padded to 3 WMMA col-tiles
#define DD   64           // input emb dim
#define BN_EPS 1e-5f
#define SCAN_B 256

typedef __attribute__((ext_vector_type(16))) __bf16 v16bf;
typedef __attribute__((ext_vector_type(8)))  float  v8f;

// ---------------- utility ----------------
__global__ void zero_f32_kernel(float* __restrict__ p, long n) {
  long i = (long)blockIdx.x * blockDim.x + threadIdx.x;
  if (i < n) p[i] = 0.0f;
}

__global__ void zero_i32_kernel(int* __restrict__ p, long n) {
  long i = (long)blockIdx.x * blockDim.x + threadIdx.x;
  if (i < n) p[i] = 0;
}

__global__ void set_i32_kernel(int* __restrict__ p, int v) { *p = v; }

// ---------------- CSR build (counting sort of edges by dst) ----------------
__global__ void count_deg_kernel(const int* __restrict__ dst, int* __restrict__ ideg, int E) {
  int e = blockIdx.x * blockDim.x + threadIdx.x;
  if (e < E) atomicAdd(&ideg[dst[e]], 1);
}

__global__ void scan_block_kernel(const int* __restrict__ ideg, int* __restrict__ bsum, int n) {
  __shared__ int sh[SCAN_B];
  int i = blockIdx.x * SCAN_B + threadIdx.x;
  sh[threadIdx.x] = (i < n) ? ideg[i] : 0;
  __syncthreads();
#pragma unroll
  for (int off = SCAN_B / 2; off > 0; off >>= 1) {
    if (threadIdx.x < off) sh[threadIdx.x] += sh[threadIdx.x + off];
    __syncthreads();
  }
  if (threadIdx.x == 0) bsum[blockIdx.x] = sh[0];
}

__global__ void scan_partials_kernel(int* __restrict__ bsum, int nb) {
  if (threadIdx.x == 0) {
    int run = 0;
    for (int b = 0; b < nb; ++b) { int v = bsum[b]; bsum[b] = run; run += v; }
  }
}

__global__ void scan_final_kernel(const int* __restrict__ ideg, const int* __restrict__ bsum,
                                  int* __restrict__ start, int* __restrict__ cursor, int n) {
  __shared__ int sh[SCAN_B];
  int i = blockIdx.x * SCAN_B + threadIdx.x;
  int v = (i < n) ? ideg[i] : 0;
  sh[threadIdx.x] = v;
  __syncthreads();
#pragma unroll
  for (int off = 1; off < SCAN_B; off <<= 1) {   // inclusive Hillis-Steele
    int t = (threadIdx.x >= off) ? sh[threadIdx.x - off] : 0;
    __syncthreads();
    sh[threadIdx.x] += t;
    __syncthreads();
  }
  if (i < n) {
    int ex = bsum[blockIdx.x] + sh[threadIdx.x] - v;   // exclusive prefix
    start[i]  = ex;
    cursor[i] = ex;
  }
}

__global__ void fill_nbr_kernel(const int* __restrict__ src, const int* __restrict__ dst,
                                int* __restrict__ cursor, int* __restrict__ nbr, int E) {
  int e = blockIdx.x * blockDim.x + threadIdx.x;
  if (e < E) {
    int p = atomicAdd(&cursor[dst[e]], 1);
    nbr[p] = src[e];
  }
}

// ---------------- neighbor mean aggregation (gather, no fp atomics) ----------------
// one wave per node; lane owns a float4 feature chunk held in registers
template <int F>
__global__ void gather_mean_kernel(const float* __restrict__ cur, const int* __restrict__ nbr,
                                   const int* __restrict__ start, float* __restrict__ agg,
                                   int n_nodes) {
  const int node = blockIdx.x * (blockDim.x >> 5) + (threadIdx.x >> 5);
  if (node >= n_nodes) return;
  const int lane = threadIdx.x & 31;
  constexpr int CH = F / 4;                 // 16 (F=64) or 32 (F=128) chunks
  const int s0 = start[node], s1 = start[node + 1];
  if (lane < CH) {
    float4 acc = make_float4(0.f, 0.f, 0.f, 0.f);
    for (int p = s0; p < s1; ++p) {
      const int nb = nbr[p];
      const float4 v = *(const float4*)(cur + (size_t)nb * F + lane * 4);
      acc.x += v.x; acc.y += v.y; acc.z += v.z; acc.w += v.w;
    }
    const float inv = 1.0f / fmaxf((float)(s1 - s0), 1.0f);
    acc.x *= inv; acc.y *= inv; acc.z *= inv; acc.w *= inv;
    *(float4*)(agg + (size_t)node * F + lane * 4) = acc;
  }
}

// ---------------- weight pre-pack: f32 row-major -> bf16 WMMA B-fragments ----------------
__global__ void pack_w_kernel(const float* __restrict__ W, __bf16* __restrict__ P,
                              int K, int ncols_mem, int ncols_pad) {
  const int nct  = ncols_pad >> 4;
  const int tile = blockIdx.x;              // kt * nct + ct
  const int kt   = tile / nct;
  const int ct   = tile - kt * nct;
  const int lane = threadIdx.x;             // 32
  const int hl   = lane >> 4;
  const int l16  = lane & 15;
  const int col  = ct * 16 + l16;
  __bf16* dst = P + ((size_t)tile * 32 + lane) * 16;
#pragma unroll
  for (int e = 0; e < 16; ++e) {
    int kb = kt * 32 + hl * 8 + (e < 8 ? e : 8 + e);
    float v = (col < ncols_mem) ? W[(size_t)kb * ncols_mem + col] : 0.0f;
    dst[e] = (__bf16)v;
  }
}

// ---------------- A-fragment loader: two contiguous 8-float runs -> 4x b128 loads ----------------
static __device__ __forceinline__ v16bf load_a_frag(const float* __restrict__ row,
                                                    int k0, int hl) {
  const float4* r4 = (const float4*)row;
  const int i0 = (k0 + hl * 8) >> 2;
  const int i1 = (k0 + 16 + hl * 8) >> 2;
  float4 p0 = r4[i0], p1 = r4[i0 + 1], p2 = r4[i1], p3 = r4[i1 + 1];
  v16bf a;
  a[0]  = (__bf16)p0.x; a[1]  = (__bf16)p0.y; a[2]  = (__bf16)p0.z; a[3]  = (__bf16)p0.w;
  a[4]  = (__bf16)p1.x; a[5]  = (__bf16)p1.y; a[6]  = (__bf16)p1.z; a[7]  = (__bf16)p1.w;
  a[8]  = (__bf16)p2.x; a[9]  = (__bf16)p2.y; a[10] = (__bf16)p2.z; a[11] = (__bf16)p2.w;
  a[12] = (__bf16)p3.x; a[13] = (__bf16)p3.y; a[14] = (__bf16)p3.z; a[15] = (__bf16)p3.w;
  return a;
}

// ---------------- fused SAGE GEMM: y = agg@Wl + cur@Wr + b ----------------
// block = 256 threads = 8 waves; block owns a 16-row tile, wave w owns cols [16w,16w+16)
template <int K>
__global__ void sage_gemm_wmma(const float* __restrict__ agg, const float* __restrict__ cur,
                               const __bf16* __restrict__ PWl, const __bf16* __restrict__ PWr,
                               const float* __restrict__ bias, float* __restrict__ y) {
  const int row0 = blockIdx.x * 16;
  const int lane = threadIdx.x & 31;
  const int wave = threadIdx.x >> 5;
  const int col0 = wave * 16;
  const int hl   = lane >> 4;
  const int l16  = lane & 15;
  const int row  = row0 + l16;
  const float* arow = agg + (size_t)row * K;
  const float* xrow = cur + (size_t)row * K;
  const v16bf* PL = (const v16bf*)PWl;
  const v16bf* PR = (const v16bf*)PWr;

  v8f c = {};
#pragma unroll
  for (int k0 = 0; k0 < K; k0 += 32) {
    v16bf a_agg = load_a_frag(arow, k0, hl);
    v16bf a_x   = load_a_frag(xrow, k0, hl);
    size_t fi = (size_t)((k0 >> 5) * (HD / 16) + wave) * 32 + lane;
    v16bf b_l = PL[fi];
    v16bf b_r = PR[fi];
    c = __builtin_amdgcn_wmma_f32_16x16x32_bf16(false, a_agg, false, b_l, (short)0, c, false, false);
    c = __builtin_amdgcn_wmma_f32_16x16x32_bf16(false, a_x,   false, b_r, (short)0, c, false, false);
  }
  const float bc = bias[col0 + l16];
#pragma unroll
  for (int r = 0; r < 8; ++r) {
    int m = row0 + r + 8 * hl;
    y[(size_t)m * HD + col0 + l16] = c[r] + bc;
  }
}

// ---------------- MLP1 GEMM: y = x@W + b (K=128, 128 cols) ----------------
__global__ void gemm128_wmma(const float* __restrict__ x, const __bf16* __restrict__ PW,
                             const float* __restrict__ bias, float* __restrict__ y) {
  const int row0 = blockIdx.x * 16;
  const int lane = threadIdx.x & 31;
  const int wave = threadIdx.x >> 5;
  const int col0 = wave * 16;
  const int hl   = lane >> 4;
  const int l16  = lane & 15;
  const float* xrow = x + (size_t)(row0 + l16) * HD;
  const v16bf* PB = (const v16bf*)PW;

  v8f c = {};
#pragma unroll
  for (int k0 = 0; k0 < HD; k0 += 32) {
    v16bf a = load_a_frag(xrow, k0, hl);
    v16bf b = PB[(size_t)((k0 >> 5) * (HD / 16) + wave) * 32 + lane];
    c = __builtin_amdgcn_wmma_f32_16x16x32_bf16(false, a, false, b, (short)0, c, false, false);
  }
  const float bc = bias[col0 + l16];
#pragma unroll
  for (int r = 0; r < 8; ++r) {
    int m = row0 + r + 8 * hl;
    y[(size_t)m * HD + col0 + l16] = c[r] + bc;
  }
}

// ---------------- BatchNorm ----------------
__global__ void bn_stats_kernel(const float* __restrict__ y, float* __restrict__ sums,
                                float* __restrict__ sumsq, int n_nodes, int rows_per_block) {
  const int c  = threadIdx.x;
  const int r0 = blockIdx.x * rows_per_block;
  const int r1 = min(r0 + rows_per_block, n_nodes);
  float s = 0.0f, q = 0.0f;
  for (int r = r0; r < r1; ++r) {
    float v = y[(size_t)r * HD + c];
    s += v; q += v * v;
  }
  atomicAdd(&sums[c], s);
  atomicAdd(&sumsq[c], q);
}

__global__ void bn_finalize_kernel(const float* __restrict__ sums, const float* __restrict__ sumsq,
                                   const float* __restrict__ g, const float* __restrict__ be,
                                   float* __restrict__ scale, float* __restrict__ shift, int n_nodes) {
  int c = threadIdx.x;
  float inv_n = 1.0f / (float)n_nodes;
  float mean = sums[c] * inv_n;
  float var  = sumsq[c] * inv_n - mean * mean;
  float sc   = g[c] * rsqrtf(var + BN_EPS);
  scale[c] = sc;
  shift[c] = be[c] - mean * sc;
}

__global__ void bn_apply_relu_kernel(const float* __restrict__ y, const float* __restrict__ scale,
                                     const float* __restrict__ shift, float* __restrict__ out, long total) {
  long i = (long)blockIdx.x * blockDim.x + threadIdx.x;
  if (i < total) {
    int c = (int)(i & (HD - 1));
    out[i] = fmaxf(y[i] * scale[c] + shift[c], 0.0f);
  }
}

// ---------------- MLP2 (128->40, padded 48) + log_softmax, fused ----------------
__global__ void head_softmax_wmma(const float* __restrict__ x, const __bf16* __restrict__ PW2,
                                  const float* __restrict__ b2, float* __restrict__ out) {
  __shared__ float lds[16 * CDP];
  const int row0 = blockIdx.x * 16;
  const int lane = threadIdx.x & 31;
  const int wave = threadIdx.x >> 5;
  const int hl   = lane >> 4;
  const int l16  = lane & 15;

  if (wave < 3) {
    const int col = wave * 16 + l16;
    const float* xrow = x + (size_t)(row0 + l16) * HD;
    const v16bf* PB = (const v16bf*)PW2;
    v8f c = {};
#pragma unroll
    for (int k0 = 0; k0 < HD; k0 += 32) {
      v16bf a = load_a_frag(xrow, k0, hl);
      v16bf b = PB[(size_t)((k0 >> 5) * (CDP / 16) + wave) * 32 + lane];
      c = __builtin_amdgcn_wmma_f32_16x16x32_bf16(false, a, false, b, (short)0, c, false, false);
    }
    const float bc = (col < CD) ? b2[col] : 0.0f;
#pragma unroll
    for (int r = 0; r < 8; ++r) {
      int m = r + 8 * hl;
      lds[m * CDP + col] = c[r] + bc;
    }
  }
  __syncthreads();
  if (threadIdx.x < 16) {
    const float* lrow = lds + threadIdx.x * CDP;
    float mx = -3.402823466e38f;
    for (int j = 0; j < CD; ++j) mx = fmaxf(mx, lrow[j]);
    float s = 0.0f;
    for (int j = 0; j < CD; ++j) s += expf(lrow[j] - mx);
    float lg = logf(s);
    float* orow = out + (size_t)(row0 + threadIdx.x) * CD;
    for (int j = 0; j < CD; ++j) orow[j] = lrow[j] - mx - lg;
  }
}

// ---------------- launch ----------------
static inline unsigned nblk(long n, int b) { return (unsigned)((n + b - 1) / b); }

extern "C" void kernel_launch(void* const* d_in, const int* in_sizes, int n_in,
                              void* d_out, int out_size, void* d_ws, size_t ws_size,
                              hipStream_t stream) {
  const float* x    = (const float*)d_in[0];
  const int*   eidx = (const int*)d_in[1];
  const int N = in_sizes[0] / DD;       // 100000 (divisible by 16)
  const int E = in_sizes[1] / 2;        // 1600000
  const int* src = eidx;
  const int* dst = eidx + E;

  const float *Wl1 = (const float*)d_in[3],  *Wr1 = (const float*)d_in[4],  *b1 = (const float*)d_in[5];
  const float *Wl2 = (const float*)d_in[6],  *Wr2 = (const float*)d_in[7],  *b2 = (const float*)d_in[8];
  const float *Wl3 = (const float*)d_in[9],  *Wr3 = (const float*)d_in[10], *b3 = (const float*)d_in[11];
  const float *g1  = (const float*)d_in[12], *be1 = (const float*)d_in[13];
  const float *g2  = (const float*)d_in[14], *be2 = (const float*)d_in[15];
  const float *g3  = (const float*)d_in[16], *be3 = (const float*)d_in[17];
  const float *Wm1 = (const float*)d_in[18], *bm1 = (const float*)d_in[19];
  const float *Wm2 = (const float*)d_in[20], *bm2 = (const float*)d_in[21];
  float* out = (float*)d_out;

  // ---------------- workspace layout (32B-aligned blocks) ----------------
  char* ws = (char*)d_ws;
  const size_t bufBytes = (size_t)N * HD * sizeof(float);
  float* B0 = (float*)(ws);
  float* B1 = (float*)(ws + bufBytes);
  float* B2 = (float*)(ws + 2 * bufBytes);
  // packed bf16 weights
  const size_t fragElems64   = (size_t)(DD / 32) * (HD / 16) * 32 * 16;   // 8192
  const size_t fragElems128  = (size_t)(HD / 32) * (HD / 16) * 32 * 16;   // 16384
  const size_t fragElemsHead = (size_t)(HD / 32) * (CDP / 16) * 32 * 16;  // 6144
  __bf16* PWl1 = (__bf16*)(ws + 3 * bufBytes);
  __bf16* PWr1 = PWl1 + fragElems64;
  __bf16* PWl2 = PWr1 + fragElems64;
  __bf16* PWr2 = PWl2 + fragElems128;
  __bf16* PWl3 = PWr2 + fragElems128;
  __bf16* PWr3 = PWl3 + fragElems128;
  __bf16* PWm1 = PWr3 + fragElems128;
  __bf16* PWm2 = PWm1 + fragElems128;
  // BN scratch
  float* sums  = (float*)(PWm2 + fragElemsHead);
  float* sumsq = sums + HD;
  float* scale = sumsq + HD;
  float* shift = scale + HD;
  // CSR scratch
  const int nbS = (int)nblk(N, SCAN_B);
  int* ideg   = (int*)(shift + HD);
  int* start  = ideg + N;          // N+1 entries
  int* cursor = start + (N + 1);
  int* bsum   = cursor + N;        // nbS entries
  int* nbr    = bsum + ((nbS + 7) & ~7);

  const long totNH = (long)N * HD;
  const int  rpb   = 256;
  (void)n_in; (void)out_size; (void)ws_size;

  // ---------------- weight packing (tiny; once per call) ----------------
  pack_w_kernel<<<(DD / 32) * (HD / 16), 32, 0, stream>>>(Wl1, PWl1, DD, HD, HD);
  pack_w_kernel<<<(DD / 32) * (HD / 16), 32, 0, stream>>>(Wr1, PWr1, DD, HD, HD);
  pack_w_kernel<<<(HD / 32) * (HD / 16), 32, 0, stream>>>(Wl2, PWl2, HD, HD, HD);
  pack_w_kernel<<<(HD / 32) * (HD / 16), 32, 0, stream>>>(Wr2, PWr2, HD, HD, HD);
  pack_w_kernel<<<(HD / 32) * (HD / 16), 32, 0, stream>>>(Wl3, PWl3, HD, HD, HD);
  pack_w_kernel<<<(HD / 32) * (HD / 16), 32, 0, stream>>>(Wr3, PWr3, HD, HD, HD);
  pack_w_kernel<<<(HD / 32) * (HD / 16), 32, 0, stream>>>(Wm1, PWm1, HD, HD, HD);
  pack_w_kernel<<<(HD / 32) * (CDP / 16), 32, 0, stream>>>(Wm2, PWm2, HD, CD, CDP);

  // ---------------- CSR build: counting sort of edges by dst ----------------
  zero_i32_kernel<<<nblk(N, 256), 256, 0, stream>>>(ideg, N);
  count_deg_kernel<<<nblk(E, 256), 256, 0, stream>>>(dst, ideg, E);
  scan_block_kernel<<<nbS, SCAN_B, 0, stream>>>(ideg, bsum, N);
  scan_partials_kernel<<<1, 32, 0, stream>>>(bsum, nbS);
  scan_final_kernel<<<nbS, SCAN_B, 0, stream>>>(ideg, bsum, start, cursor, N);
  set_i32_kernel<<<1, 1, 0, stream>>>(start + N, E);
  fill_nbr_kernel<<<nblk(E, 256), 256, 0, stream>>>(src, dst, cursor, nbr, E);

  // ---------------- Layer 1 (F=64) ----------------
  gather_mean_kernel<DD><<<nblk(N, 8), 256, 0, stream>>>(x, nbr, start, B0, N);
  sage_gemm_wmma<DD><<<N / 16, 256, 0, stream>>>(B0, x, PWl1, PWr1, b1, B2);
  zero_f32_kernel<<<1, 256, 0, stream>>>(sums, 2 * HD);
  bn_stats_kernel<<<nblk(N, rpb), HD, 0, stream>>>(B2, sums, sumsq, N, rpb);
  bn_finalize_kernel<<<1, HD, 0, stream>>>(sums, sumsq, g1, be1, scale, shift, N);
  bn_apply_relu_kernel<<<nblk(totNH, 256), 256, 0, stream>>>(B2, scale, shift, B1, totNH);

  // ---------------- Layer 2 (F=128) ----------------
  gather_mean_kernel<HD><<<nblk(N, 8), 256, 0, stream>>>(B1, nbr, start, B0, N);
  sage_gemm_wmma<HD><<<N / 16, 256, 0, stream>>>(B0, B1, PWl2, PWr2, b2, B2);
  zero_f32_kernel<<<1, 256, 0, stream>>>(sums, 2 * HD);
  bn_stats_kernel<<<nblk(N, rpb), HD, 0, stream>>>(B2, sums, sumsq, N, rpb);
  bn_finalize_kernel<<<1, HD, 0, stream>>>(sums, sumsq, g2, be2, scale, shift, N);
  bn_apply_relu_kernel<<<nblk(totNH, 256), 256, 0, stream>>>(B2, scale, shift, B0, totNH);

  // ---------------- Layer 3 (F=128) ----------------
  gather_mean_kernel<HD><<<nblk(N, 8), 256, 0, stream>>>(B0, nbr, start, B1, N);
  sage_gemm_wmma<HD><<<N / 16, 256, 0, stream>>>(B1, B0, PWl3, PWr3, b3, B2);
  zero_f32_kernel<<<1, 256, 0, stream>>>(sums, 2 * HD);
  bn_stats_kernel<<<nblk(N, rpb), HD, 0, stream>>>(B2, sums, sumsq, N, rpb);
  bn_finalize_kernel<<<1, HD, 0, stream>>>(sums, sumsq, g3, be3, scale, shift, N);
  bn_apply_relu_kernel<<<nblk(totNH, 256), 256, 0, stream>>>(B2, scale, shift, B1, totNH);

  // ---------------- MLP head ----------------
  gemm128_wmma<<<N / 16, 256, 0, stream>>>(B1, PWm1, bm1, B0);
  head_softmax_wmma<<<N / 16, 128, 0, stream>>>(B0, PWm2, bm2, out);
}
